// BiMambaLayer_62216896250084
// MI455X (gfx1250) — matrix operand order repaired and checked by reading, not compiled, not run on hardware
//
#include <hip/hip_runtime.h>

typedef __attribute__((ext_vector_type(16))) __bf16 v16bf;
typedef __attribute__((ext_vector_type(8)))  float  v8f;
typedef unsigned short u16;
typedef unsigned int   u32;

typedef int v4i __attribute__((vector_size(16)));
typedef __attribute__((address_space(1))) v4i* gptr_v4i;
typedef __attribute__((address_space(3))) v4i* lptr_v4i;

#define DMODEL   1024
#define DSTATE   16
#define DINNER   2048
#define NHEADS   32
#define HEADDIM  64
#define CONVDIM  2080
#define DIP      4160
#define BATCH    2
#define SEQ      2048
#define BLROWS   (BATCH * SEQ)       // 4096
#define RMS_EPS  1e-5f

struct U4 { u32 a, b, c, d; };
struct U8 { U4 lo, hi; };

#if __has_builtin(__builtin_amdgcn_global_load_async_to_lds_b128)
#define ASYNC_LDS 1
#else
#define ASYNC_LDS 0
#endif

static __device__ inline v16bf load_frag(const u16* __restrict__ p0,
                                         const u16* __restrict__ p1) {
  U8 u;
  u.lo = *(const U4*)p0;
  u.hi = *(const U4*)p1;
  return __builtin_bit_cast(v16bf, u);
}

static __device__ inline u16 f32_to_bf16(float f) {
  u32 u = __builtin_bit_cast(u32, f);
  u32 r = (u + 0x7FFFu + ((u >> 16) & 1u)) >> 16;   // round-to-nearest-even
  return (u16)r;
}

static __device__ inline float sigmoidf_(float x) { return 1.0f / (1.0f + __expf(-x)); }

// 16-byte global -> LDS copy (async on CDNA5, sync fallback otherwise)
static __device__ inline void cp_g2l_b128(u16* dst, const u16* src) {
#if ASYNC_LDS
  __builtin_amdgcn_global_load_async_to_lds_b128(
      (gptr_v4i)(size_t)src,
      (lptr_v4i)(u32)(size_t)dst,
      0, 0);
#else
  *(U4*)dst = *(const U4*)src;
#endif
}

template <int N>
static __device__ inline void wait_async_le() {
#if ASYNC_LDS
#if __has_builtin(__builtin_amdgcn_s_wait_asynccnt)
  __builtin_amdgcn_s_wait_asynccnt((unsigned short)N);
#else
  asm volatile("s_wait_asynccnt %0" :: "i"(N) : "memory");
#endif
#endif
}

// ---------------------------------------------------------------------------
// f32 -> bf16 conversion kernels
// ---------------------------------------------------------------------------
__global__ void cvt_bf16_kernel(const float* __restrict__ in, u16* __restrict__ out, int n) {
  int i = blockIdx.x * blockDim.x + threadIdx.x;
  if (i < n) out[i] = f32_to_bf16(in[i]);
}

// out[r*ostride + c] = bf16(in[r*ncols + c])  (used to pack h_fwd/h_bwd concat)
__global__ void cvt_bf16_strided_kernel(const float* __restrict__ in, u16* __restrict__ out,
                                        int ncols, int ostride, int n) {
  int i = blockIdx.x * blockDim.x + threadIdx.x;
  if (i < n) {
    int r = i / ncols, c = i - r * ncols;
    out[(size_t)r * ostride + c] = f32_to_bf16(in[i]);
  }
}

// ---------------------------------------------------------------------------
// WMMA GEMM:  C[M x N] (f32) = A[M x K] (bf16, row-major) * W[N x K]^T (bf16)
// 256 threads = 8 waves. Block tile = 128(M) x 64(N); wave tile = 16(M) x 64(N).
// The 64-column B tile is staged in LDS in 64-deep K chunks with async
// global->LDS copies, double-buffered; steady-state loop is branch-free.
// ---------------------------------------------------------------------------
__global__ __launch_bounds__(256) void gemm_bf16_wmma_kernel(
    const u16* __restrict__ A, const u16* __restrict__ W, float* __restrict__ C,
    int M, int N, int K) {
  __shared__ u16 bstage[2][64 * 64];         // [buf][col][k] : 2 x 8KB

  const int tid  = threadIdx.x;
  const int lane = tid & 31;
  const int wave = tid >> 5;
  const int nt64 = N >> 6;
  const int bm   = blockIdx.x / nt64;        // 128-row block index
  const int n64  = blockIdx.x - bm * nt64;   // 64-col block index
  const int row  = lane & 15;
  const int hk   = lane >> 4;                // half-wave selects K sub-range

  // --- cooperative B-stage copy mapping: thread -> (col, 2 x 16B of K) -----
  const int colc  = tid >> 2;                // 0..63
  const int koffc = (tid & 3) << 3;          // 0,8,16,24 (bf16 elems)
  const u16* __restrict__ wsrc = W + (size_t)(n64 * 64 + colc) * K + koffc;
  u16* const lbuf0 = &bstage[0][colc * 64 + koffc];
  u16* const lbuf1 = &bstage[1][colc * 64 + koffc];

  // --- per-wave A fragment source ------------------------------------------
  // lanes 0-15: row M=row, K = k+[0..15]; lanes 16-31: K = k+[16..31]
  const u16* __restrict__ arow =
      A + (size_t)(bm * 128 + wave * 16 + row) * K + hk * 16;

  v8f acc0 = {}, acc1 = {}, acc2 = {}, acc3 = {};

  // one double-rate (2 x 16x16x32) WMMA stage over a 64-deep K chunk
  auto compute_stage = [&](int cur, int kbase) {
    const u16* __restrict__ bb = &bstage[cur][0];
#pragma unroll
    for (int ks = 0; ks < 2; ++ks) {
      const int k = kbase + ks * 32;
      __builtin_prefetch(arow + k + 256, 0, 1);
      v16bf a = load_frag(arow + k, arow + k + 8);
      // B fragment: lane holds column (j*16+row), 16 contiguous local-K vals
      const u16* bk = bb + row * 64 + ks * 32 + hk * 16;
      v16bf b0 = load_frag(bk,             bk + 8);
      v16bf b1 = load_frag(bk + 16 * 64,   bk + 16 * 64 + 8);
      v16bf b2 = load_frag(bk + 32 * 64,   bk + 32 * 64 + 8);
      v16bf b3 = load_frag(bk + 48 * 64,   bk + 48 * 64 + 8);
      acc0 = __builtin_amdgcn_wmma_f32_16x16x32_bf16(false, a, false, b0, (short)0, acc0, false, false);
      acc1 = __builtin_amdgcn_wmma_f32_16x16x32_bf16(false, a, false, b1, (short)0, acc1, false, false);
      acc2 = __builtin_amdgcn_wmma_f32_16x16x32_bf16(false, a, false, b2, (short)0, acc2, false, false);
      acc3 = __builtin_amdgcn_wmma_f32_16x16x32_bf16(false, a, false, b3, (short)0, acc3, false, false);
    }
  };

  const int nk = K >> 6;                     // 64-deep K chunks (>= 2 here)
  // prologue: stage chunk 0 into buffer 0
  cp_g2l_b128(lbuf0,      wsrc);
  cp_g2l_b128(lbuf0 + 32, wsrc + 32);

  // steady state: always prefetching the next chunk (branch-free body)
  for (int kc = 0; kc < nk - 1; ++kc) {
    const int cur = kc & 1;
    const u16* s = wsrc + (size_t)(kc + 1) * 64;
    u16* d = cur ? lbuf0 : lbuf1;
    cp_g2l_b128(d,      s);
    cp_g2l_b128(d + 32, s + 32);
    wait_async_le<2>();                      // chunk kc's copies complete
    __syncthreads();                         // stage visible to all waves
    compute_stage(cur, kc * 64);
    __syncthreads();                         // waves done reading this buffer
  }
  // epilogue: last chunk
  wait_async_le<0>();
  __syncthreads();
  compute_stage((nk - 1) & 1, (nk - 1) * 64);

  // D layout: VGPR v, lanes 0-15 -> (M=v, N=lane); lanes 16-31 -> (M=v+8, N=lane-16)
  const int crow0 = bm * 128 + wave * 16 + hk * 8;
  float* __restrict__ cbase = C + (size_t)crow0 * N + n64 * 64 + row;
#pragma unroll
  for (int v = 0; v < 8; ++v) {
    float* cr = cbase + (size_t)v * N;
    cr[0]  = acc0[v];
    cr[16] = acc1[v];
    cr[32] = acc2[v];
    cr[48] = acc3[v];
  }
}

// ---------------------------------------------------------------------------
// Depthwise causal conv (D_CONV=4) + bias + SiLU over the xBC slice of zxbcdt.
// rev=1: conv over time-reversed sequence, written at original positions.
// ---------------------------------------------------------------------------
__global__ void conv_silu_kernel(const float* __restrict__ zx,
                                 const float* __restrict__ cw,   // (CONVDIM,1,4) flat
                                 const float* __restrict__ cb,
                                 float* __restrict__ xbc, int rev) {
  int i = blockIdx.x * blockDim.x + threadIdx.x;
  const int total = BLROWS * CONVDIM;
  if (i >= total) return;
  int c  = i % CONVDIM;
  int bt = i / CONVDIM;
  int t  = bt % SEQ;
  int b  = bt / SEQ;
  const float* __restrict__ base = zx + (size_t)b * SEQ * DIP + DINNER + c;  // stride DIP in t
  float s = cb[c];
  if (!rev) {
#pragma unroll
    for (int j = 0; j < 4; ++j) {
      int tt = t - 3 + j;
      if (tt >= 0) s += cw[c * 4 + j] * base[(size_t)tt * DIP];
    }
  } else {
#pragma unroll
    for (int d = 0; d < 4; ++d) {
      int tt = t + d;
      if (tt < SEQ) s += cw[c * 4 + 3 - d] * base[(size_t)tt * DIP];
    }
  }
  xbc[(size_t)bt * CONVDIM + c] = s * sigmoidf_(s);
}

// ---------------------------------------------------------------------------
// Sequential SSM scan. One wave per (batch, head); each lane owns 2 head-dim
// rows x 16 state columns in registers. rev=1 scans t = L-1..0.
// Writes y = scan_out + Dp * x  into (B,L,H,P) f32.
// ---------------------------------------------------------------------------
__global__ __launch_bounds__(32) void ssm_scan_kernel(
    const float* __restrict__ xbc, const float* __restrict__ zx,
    const float* __restrict__ dt_bias, const float* __restrict__ A_log,
    const float* __restrict__ Dp, float* __restrict__ y, int rev) {
  const int b = blockIdx.x / NHEADS;
  const int h = blockIdx.x % NHEADS;
  const int lane = threadIdx.x;
  const float Ah  = -__expf(A_log[h]);
  const float dtb = dt_bias[h];
  const float Dv  = Dp[h];

  float st0[DSTATE], st1[DSTATE];
#pragma unroll
  for (int n = 0; n < DSTATE; ++n) { st0[n] = 0.f; st1[n] = 0.f; }

  for (int step = 0; step < SEQ; ++step) {
    const int t = rev ? (SEQ - 1 - step) : step;
    const size_t rowi = (size_t)b * SEQ + t;
    float dtr = zx[rowi * DIP + (DINNER + CONVDIM) + h] + dtb;
    float dt  = (dtr > 20.f) ? dtr : log1pf(__expf(dtr));     // softplus
    float dA  = __expf(dt * Ah);
    const float* __restrict__ xr = xbc + rowi * CONVDIM;
    float Bv[DSTATE], Cv[DSTATE];
#pragma unroll
    for (int n = 0; n < DSTATE; ++n) {
      Bv[n] = xr[DINNER + n];
      Cv[n] = xr[DINNER + DSTATE + n];
    }
    const float x0 = xr[h * HEADDIM + lane * 2 + 0];
    const float x1 = xr[h * HEADDIM + lane * 2 + 1];
    const float dtx0 = dt * x0, dtx1 = dt * x1;
    float y0 = 0.f, y1 = 0.f;
#pragma unroll
    for (int n = 0; n < DSTATE; ++n) {
      st0[n] = st0[n] * dA + Bv[n] * dtx0;
      st1[n] = st1[n] * dA + Bv[n] * dtx1;
      y0 += Cv[n] * st0[n];
      y1 += Cv[n] * st1[n];
    }
    float* __restrict__ yr = y + rowi * DINNER + h * HEADDIM + lane * 2;
    yr[0] = y0 + Dv * x0;
    yr[1] = y1 + Dv * x1;
  }
}

// ---------------------------------------------------------------------------
// y = y * silu(z); RMSNorm over D_INNER with norm_w; emit bf16.
// One 256-thread block per (b,l) row (2048 elements -> 8 per thread).
// ---------------------------------------------------------------------------
__global__ __launch_bounds__(256) void gate_rmsnorm_kernel(
    const float* __restrict__ y, const float* __restrict__ zx,
    const float* __restrict__ nw, u16* __restrict__ ybf) {
  const int rowid = blockIdx.x;
  const float* __restrict__ yr = y  + (size_t)rowid * DINNER;
  const float* __restrict__ zr = zx + (size_t)rowid * DIP;      // z = slice [0, DINNER)
  __shared__ float red[8];
  float vals[8];
  float ssum = 0.f;
#pragma unroll
  for (int i = 0; i < 8; ++i) {
    int idx = threadIdx.x + i * 256;
    float z = zr[idx];
    float g = yr[idx] * (z * sigmoidf_(z));
    vals[i] = g;
    ssum += g * g;
  }
#pragma unroll
  for (int off = 16; off > 0; off >>= 1) ssum += __shfl_down(ssum, off, 32);
  if ((threadIdx.x & 31) == 0) red[threadIdx.x >> 5] = ssum;
  __syncthreads();
  float tot = 0.f;
#pragma unroll
  for (int j = 0; j < 8; ++j) tot += red[j];
  const float scale = rsqrtf(tot * (1.0f / DINNER) + RMS_EPS);
#pragma unroll
  for (int i = 0; i < 8; ++i) {
    int idx = threadIdx.x + i * 256;
    ybf[(size_t)rowid * DINNER + idx] = f32_to_bf16(vals[i] * scale * nw[idx]);
  }
}

// ---------------------------------------------------------------------------
// out = x + mamba_out * layer_scale
// ---------------------------------------------------------------------------
__global__ void final_residual_kernel(const float* __restrict__ x,
                                      const float* __restrict__ mo,
                                      const float* __restrict__ scale,
                                      float* __restrict__ out, int n) {
  int i = blockIdx.x * blockDim.x + threadIdx.x;
  if (i < n) out[i] = x[i] + mo[i] * scale[i % DMODEL];
}

// ---------------------------------------------------------------------------
extern "C" void kernel_launch(void* const* d_in, const int* in_sizes, int n_in,
                              void* d_out, int out_size, void* d_ws, size_t ws_size,
                              hipStream_t stream) {
  (void)in_sizes; (void)n_in; (void)out_size; (void)ws_size;
  const float* x_in        = (const float*)d_in[0];
  const float* in_proj_w[2]  = { (const float*)d_in[1],  (const float*)d_in[9]  };
  const float* conv_w[2]     = { (const float*)d_in[2],  (const float*)d_in[10] };
  const float* conv_b[2]     = { (const float*)d_in[3],  (const float*)d_in[11] };
  const float* dt_bias[2]    = { (const float*)d_in[4],  (const float*)d_in[12] };
  const float* A_log[2]      = { (const float*)d_in[5],  (const float*)d_in[13] };
  const float* Dp[2]         = { (const float*)d_in[6],  (const float*)d_in[14] };
  const float* norm_w[2]     = { (const float*)d_in[7],  (const float*)d_in[15] };
  const float* out_proj_w[2] = { (const float*)d_in[8],  (const float*)d_in[16] };
  const float* layer_w     = (const float*)d_in[17];
  const float* layer_scale = (const float*)d_in[18];
  float* out = (float*)d_out;

  char* ws = (char*)d_ws;
  size_t off = 0;
  auto wsAlloc = [&](size_t bytes) -> void* {
    void* p = ws + off;
    off += (bytes + 255) & ~(size_t)255;
    return p;
  };
  u16*   xbf     = (u16*)  wsAlloc((size_t)BLROWS * DMODEL * 2);
  u16*   w_in    = (u16*)  wsAlloc((size_t)DIP    * DMODEL * 2);   // reused fwd/bwd
  u16*   w_out   = (u16*)  wsAlloc((size_t)DMODEL * DINNER * 2);   // reused fwd/bwd
  u16*   w_lay   = (u16*)  wsAlloc((size_t)DMODEL * DINNER * 2);
  float* zx      = (float*)wsAlloc((size_t)BLROWS * DIP    * 4);   // reused fwd/bwd
  float* xbc     = (float*)wsAlloc((size_t)BLROWS * CONVDIM* 4);   // reused
  float* ybuf    = (float*)wsAlloc((size_t)BLROWS * DINNER * 4);   // reused
  u16*   ybf     = (u16*)  wsAlloc((size_t)BLROWS * DINNER * 2);   // reused
  float* hdir0   = (float*)wsAlloc((size_t)BLROWS * DMODEL * 4);
  float* hdir1   = (float*)wsAlloc((size_t)BLROWS * DMODEL * 4);
  u16*   hbi     = (u16*)  wsAlloc((size_t)BLROWS * DINNER * 2);
  float* mo      = (float*)wsAlloc((size_t)BLROWS * DMODEL * 4);
  float* hdir[2] = { hdir0, hdir1 };

  const int CT = 256;
  auto cdiv = [](long long a, long long b) { return (int)((a + b - 1) / b); };

  // x -> bf16, layer weight -> bf16
  {
    int n = BLROWS * DMODEL;
    cvt_bf16_kernel<<<cdiv(n, CT), CT, 0, stream>>>(x_in, xbf, n);
    int nl = DMODEL * DINNER;
    cvt_bf16_kernel<<<cdiv(nl, CT), CT, 0, stream>>>(layer_w, w_lay, nl);
  }

  for (int dir = 0; dir < 2; ++dir) {
    // in_proj weight -> bf16 (buffer reused)
    {
      int n = DIP * DMODEL;
      cvt_bf16_kernel<<<cdiv(n, CT), CT, 0, stream>>>(in_proj_w[dir], w_in, n);
    }
    // zxbcdt = x @ W_in^T  : M=4096, N=4160, K=1024
    {
      int blocks = (BLROWS / 128) * (DIP / 64);
      gemm_bf16_wmma_kernel<<<blocks, 256, 0, stream>>>(xbf, w_in, zx,
                                                        BLROWS, DIP, DMODEL);
    }
    // depthwise conv + silu
    {
      int n = BLROWS * CONVDIM;
      conv_silu_kernel<<<cdiv(n, CT), CT, 0, stream>>>(zx, conv_w[dir], conv_b[dir],
                                                       xbc, dir);
    }
    // sequential SSM scan (includes Dp * x skip)
    ssm_scan_kernel<<<BATCH * NHEADS, 32, 0, stream>>>(xbc, zx, dt_bias[dir],
                                                       A_log[dir], Dp[dir], ybuf, dir);
    // gate with silu(z) + RMSNorm, emit bf16
    gate_rmsnorm_kernel<<<BLROWS, 256, 0, stream>>>(ybuf, zx, norm_w[dir], ybf);
    // out_proj weight -> bf16, then h_dir = y @ W_out^T : M=4096, N=1024, K=2048
    {
      int n = DMODEL * DINNER;
      cvt_bf16_kernel<<<cdiv(n, CT), CT, 0, stream>>>(out_proj_w[dir], w_out, n);
      int blocks = (BLROWS / 128) * (DMODEL / 64);
      gemm_bf16_wmma_kernel<<<blocks, 256, 0, stream>>>(ybf, w_out, hdir[dir],
                                                        BLROWS, DMODEL, DINNER);
    }
  }

  // pack [h_fwd | h_bwd] -> bf16 (B,L,2048)
  {
    int n = BLROWS * DMODEL;
    cvt_bf16_strided_kernel<<<cdiv(n, CT), CT, 0, stream>>>(hdir0, hbi, DMODEL, DINNER, n);
    cvt_bf16_strided_kernel<<<cdiv(n, CT), CT, 0, stream>>>(hdir1, hbi + DMODEL, DMODEL, DINNER, n);
  }
  // mamba_out = hbi @ layer_w^T : M=4096, N=1024, K=2048
  {
    int blocks = (BLROWS / 128) * (DMODEL / 64);
    gemm_bf16_wmma_kernel<<<blocks, 256, 0, stream>>>(hbi, w_lay, mo,
                                                      BLROWS, DMODEL, DINNER);
  }
  // out = x + mamba_out * layer_scale
  {
    int n = BLROWS * DMODEL;
    final_residual_kernel<<<cdiv(n, CT), CT, 0, stream>>>(x_in, mo, layer_scale, out, n);
  }
}
